// MetaEmbedding_39505109188818
// MI455X (gfx1250) — compile-verified, matrix operand order checked by reading, not run on hardware
//
#include <hip/hip_runtime.h>
#include <math.h>

// ---------------------------------------------------------------------------
// MetaEmbedding forward for MI455X (gfx1250, wave32, WMMA).
// All heavy GEMMs run through one bf16-WMMA NT-GEMM kernel
// (A[M,K] row-major  x  Bt[N,K] row-major -> C[M,N], f32 accumulate),
// software-pipelined: register staging + double-buffered LDS, 1 barrier/step.
// ---------------------------------------------------------------------------

#define TM  128   // M tile per workgroup
#define TN  128   // N tile per workgroup
#define TK  32    // K step (one bf16 WMMA K)
#define TKP 40    // padded LDS row (32 + 8 bf16) -> 80B stride, conflict-free frags

typedef __attribute__((ext_vector_type(8)))  __bf16 v8bf;
typedef __attribute__((ext_vector_type(16))) __bf16 v16bf;
typedef __attribute__((ext_vector_type(8)))  float  v8f;

// Load one 16x32 bf16 fragment from an LDS tile (row-major, TKP stride).
// Layout per CDNA5 ISA 7.12.2: lane<16 -> row=lane, K chunks {0..7,16..23};
// lane>=16 -> row=lane-16, K chunks {8..15,24..31}.
__device__ __forceinline__ v16bf frag_from_lds(const __bf16* tile, int rowBase, int lane) {
  const int r  = rowBase + (lane & 15);
  const int k0 = (lane >> 4) << 3;              // 0 or 8
  const __bf16* p = tile + r * TKP + k0;
  v8bf lo = *(const v8bf*)(p);                  // K = k0 .. k0+7
  v8bf hi = *(const v8bf*)(p + 16);             // K = k0+16 .. k0+23
  return __builtin_shufflevector(lo, hi, 0,1,2,3,4,5,6,7,8,9,10,11,12,13,14,15);
}

__device__ __forceinline__ void cvt_store16(__bf16* q, const float4* f) {
  v8bf lo, hi;
  lo[0]=(__bf16)f[0].x; lo[1]=(__bf16)f[0].y; lo[2]=(__bf16)f[0].z; lo[3]=(__bf16)f[0].w;
  lo[4]=(__bf16)f[1].x; lo[5]=(__bf16)f[1].y; lo[6]=(__bf16)f[1].z; lo[7]=(__bf16)f[1].w;
  hi[0]=(__bf16)f[2].x; hi[1]=(__bf16)f[2].y; hi[2]=(__bf16)f[2].z; hi[3]=(__bf16)f[2].w;
  hi[4]=(__bf16)f[3].x; hi[5]=(__bf16)f[3].y; hi[6]=(__bf16)f[3].z; hi[7]=(__bf16)f[3].w;
  *(v8bf*)(q)     = lo;
  *(v8bf*)(q + 8) = hi;
}

// C[M,N] = A[M,K] @ Bt[N,K]^T (+bias). M must be a multiple of TM (M=512 here).
// grid.x = M/TM (fast dim so concurrent blocks share the same B strip in L2),
// grid.y = ceil(N/TN). 256 threads = 8 waves; wave grid 2(M) x 4(N);
// each wave owns a 64x32 patch = 4x2 WMMA accumulators.
__global__ __launch_bounds__(256)
void gemm_nt_wmma(const float* __restrict__ A, int lda,
                  const float* __restrict__ Bt, int ldb,
                  const float* __restrict__ bias,
                  float* __restrict__ C, int ldc,
                  int N, int K)
{
  __shared__ __bf16 As[2][TM * TKP];
  __shared__ __bf16 Bs[2][TN * TKP];

  const int m0   = blockIdx.x * TM;
  const int n0   = blockIdx.y * TN;
  const int t    = threadIdx.x;
  const int lane = t & 31;
  const int w    = t >> 5;
  const int wm   = (w & 1) * 64;        // wave M offset in tile
  const int wn   = (w >> 1) * 32;       // wave N offset in tile
  const int lr   = t >> 1;              // tile-fill row     (0..127)
  const int lc   = (t & 1) << 4;        // tile-fill K chunk (0 or 16)

  const v8f vzero = {0.f,0.f,0.f,0.f,0.f,0.f,0.f,0.f};
  v8f acc[4][2];
#pragma unroll
  for (int i = 0; i < 4; ++i)
#pragma unroll
    for (int j = 0; j < 2; ++j) acc[i][j] = vzero;

  // Per-thread streaming pointers for the cooperative tile fill.
  const float* aptr = A  + (size_t)(m0 + lr) * lda + lc;   // rows always valid
  const float* bptr = Bt + (size_t)(n0 + lr) * ldb + lc;
  const bool   bvalid = (n0 + lr) < N;

  float4 ra[4], rb[4];
  const float4 f4zero{};
  rb[0] = f4zero; rb[1] = f4zero; rb[2] = f4zero; rb[3] = f4zero;

  auto fetch = [&](int kt) {
    const float4* pa = (const float4*)(aptr + kt);
    ra[0] = pa[0]; ra[1] = pa[1]; ra[2] = pa[2]; ra[3] = pa[3];
    if (bvalid) {
      const float4* pb = (const float4*)(bptr + kt);
      rb[0] = pb[0]; rb[1] = pb[1]; rb[2] = pb[2]; rb[3] = pb[3];
      if (kt + 2 * TK < K) __builtin_prefetch(bptr + kt + 2 * TK, 0, 0);
    }
  };

  fetch(0);

  int buf = 0;
  for (int kt = 0; kt < K; kt += TK, buf ^= 1) {
    // Drain staged registers into this step's LDS buffer (waits on loadcnt
    // of the fetch issued last iteration — that latency was hidden by WMMA).
    cvt_store16(&As[buf][lr * TKP + lc], ra);
    cvt_store16(&Bs[buf][lr * TKP + lc], rb);

    // Issue next tile's global loads BEFORE computing: overlaps HBM latency
    // with the WMMA block below.
    if (kt + TK < K) fetch(kt + TK);

    __syncthreads();   // single barrier per step (double-buffer hazard logic)

    v16bf a[4], b[2];
#pragma unroll
    for (int i = 0; i < 4; ++i) a[i] = frag_from_lds(As[buf], wm + i * 16, lane);
#pragma unroll
    for (int j = 0; j < 2; ++j) b[j] = frag_from_lds(Bs[buf], wn + j * 16, lane);
#pragma unroll
    for (int i = 0; i < 4; ++i)
#pragma unroll
      for (int j = 0; j < 2; ++j)
        acc[i][j] = __builtin_amdgcn_wmma_f32_16x16x32_bf16(
            false, a[i], false, b[j], (short)0, acc[i][j], false, false);
  }

  // Epilogue. C/D layout: VGPR r, lane l -> (M = r + 8*(l>=16), N = l&15).
#pragma unroll
  for (int j = 0; j < 2; ++j) {
    const int col = n0 + wn + j * 16 + (lane & 15);
    if (col >= N) continue;
    const float bv = bias ? bias[col] : 0.f;
#pragma unroll
    for (int i = 0; i < 4; ++i) {
      const int row = m0 + wm + i * 16 + ((lane >> 4) << 3);
#pragma unroll
      for (int r = 0; r < 8; ++r)
        C[(size_t)(row + r) * ldc + col] = acc[i][j][r] + bv;
    }
  }
}

// ---------------------------------------------------------------------------
// Auxiliary kernels
// ---------------------------------------------------------------------------

// out[row] = sum(A[row, :]^2), one block (256 thr) per row
__global__ __launch_bounds__(256)
void rowsumsq_kernel(const float* __restrict__ A, int ncols, int ld,
                     float* __restrict__ out) {
  __shared__ float red[256];
  const int tid = threadIdx.x;
  const float* row = A + (size_t)blockIdx.x * ld;
  float s = 0.f;
  for (int c = tid; c < ncols; c += 256) { float v = row[c]; s += v * v; }
  red[tid] = s; __syncthreads();
  for (int off = 128; off > 0; off >>= 1) {
    if (tid < off) red[tid] += red[tid + off];
    __syncthreads();
  }
  if (tid == 0) out[blockIdx.x] = red[0];
}

// reach[b] = SCALE / sqrt(max(min_c(x2[b] - 2*S1[b,c] + c2[c]), 0))
__global__ __launch_bounds__(256)
void min_reach_kernel(const float* __restrict__ S1, int n, int ld,
                      const float* __restrict__ x2, const float* __restrict__ c2,
                      float* __restrict__ reach) {
  __shared__ float red[256];
  const int tid = threadIdx.x;
  const int b = blockIdx.x;
  const float xb = x2[b];
  const float* row = S1 + (size_t)b * ld;
  float m = INFINITY;
  for (int c = tid; c < n; c += 256)
    m = fminf(m, xb - 2.f * row[c] + c2[c]);
  red[tid] = m; __syncthreads();
  for (int off = 128; off > 0; off >>= 1) {
    if (tid < off) red[tid] = fminf(red[tid], red[tid + off]);
    __syncthreads();
  }
  if (tid == 0) reach[b] = 10.0f / sqrtf(fmaxf(red[0], 0.f));
}

// Row softmax S[b, 0..n) -> P[b, 0..ldp) with zero padding n..ldp
__global__ __launch_bounds__(256)
void softmax_rows_kernel(const float* __restrict__ S, int n, int ldS,
                         float* __restrict__ P, int ldp) {
  __shared__ float red[256];
  const int tid = threadIdx.x;
  const int b = blockIdx.x;
  const float* row = S + (size_t)b * ldS;
  float* prow = P + (size_t)b * ldp;

  float m = -INFINITY;
  for (int c = tid; c < n; c += 256) m = fmaxf(m, row[c]);
  red[tid] = m; __syncthreads();
  for (int off = 128; off > 0; off >>= 1) {
    if (tid < off) red[tid] = fmaxf(red[tid], red[tid + off]);
    __syncthreads();
  }
  m = red[0]; __syncthreads();

  float s = 0.f;
  for (int c = tid; c < ldp; c += 256) {
    float e = (c < n) ? __expf(row[c] - m) : 0.f;
    prow[c] = e; s += e;
  }
  red[tid] = s; __syncthreads();
  for (int off = 128; off > 0; off >>= 1) {
    if (tid < off) red[tid] += red[tid + off];
    __syncthreads();
  }
  const float inv = 1.f / red[0];
  __syncthreads();
  for (int c = tid; c < n; c += 256) prow[c] *= inv;   // pad stays 0
}

// out[fp, c] = (c < C) ? in[c, fp] : 0   (K-padded transpose, ldo = padded C)
__global__ __launch_bounds__(256)
void transpose_pad_kernel(const float* __restrict__ in, int C, int F,
                          float* __restrict__ out, int ldo) {
  __shared__ float tile[32][33];
  const int c0 = blockIdx.x * 32;   // padded-C block
  const int f0 = blockIdx.y * 32;
  const int tx = threadIdx.x & 31;
  const int ty = threadIdx.x >> 5;  // 0..7
#pragma unroll
  for (int yy = 0; yy < 32; yy += 8) {
    const int c = c0 + ty + yy;
    tile[ty + yy][tx] = (c < C) ? in[(size_t)c * F + f0 + tx] : 0.f;
  }
  __syncthreads();
#pragma unroll
  for (int yy = 0; yy < 32; yy += 8) {
    const int f = f0 + ty + yy;
    out[(size_t)f * ldo + c0 + tx] = tile[tx][ty + yy];
  }
}

// infused = tanh(Ssel)*mem ; fused = reach[b]*(x + infused); emit x / infused outputs
__global__ __launch_bounds__(256)
void fuse_kernel(const float* __restrict__ x, const float* __restrict__ Ssel,
                 const float* __restrict__ mem, const float* __restrict__ reach,
                 float* __restrict__ fused, float* __restrict__ out_x,
                 float* __restrict__ out_inf, int F, int total) {
  const int i = blockIdx.x * 256 + threadIdx.x;
  if (i >= total) return;
  const int b = i / F;
  const float xv  = x[i];
  const float sel = tanhf(Ssel[i]);
  const float inf_ = sel * mem[i];
  fused[i]   = reach[b] * (xv + inf_);
  out_x[i]   = xv;
  out_inf[i] = inf_;
}

// ---------------------------------------------------------------------------
// Launch
// ---------------------------------------------------------------------------

static inline void launch_gemm(const float* A, int lda, const float* Bt, int ldb,
                               const float* bias, float* C, int ldc,
                               int M, int N, int K, hipStream_t s) {
  dim3 grid(M / TM, (N + TN - 1) / TN);
  gemm_nt_wmma<<<grid, 256, 0, s>>>(A, lda, Bt, ldb, bias, C, ldc, N, K);
}

extern "C" void kernel_launch(void* const* d_in, const int* in_sizes, int n_in,
                              void* d_out, int out_size, void* d_ws, size_t ws_size,
                              hipStream_t stream) {
  (void)in_sizes; (void)n_in; (void)out_size; (void)ws_size;

  constexpr int B = 512, F = 2048, C = 1000, CP = 1024, NCLS = 50030;

  const float* x         = (const float*)d_in[0];
  const float* centroids = (const float*)d_in[1];
  const float* W_hall    = (const float*)d_in[2];
  const float* b_hall    = (const float*)d_in[3];
  const float* W_sel     = (const float*)d_in[4];
  const float* b_sel     = (const float*)d_in[5];
  const float* W_cls     = (const float*)d_in[6];
  const float* b_cls     = (const float*)d_in[7];

  float* logits  = (float*)d_out;                       // [B, NCLS]
  float* out_x   = logits + (size_t)B * NCLS;           // [B, F]
  float* out_inf = out_x  + (size_t)B * F;              // [B, F]

  // Workspace layout (floats)
  float* ws    = (float*)d_ws;
  float* centT = ws;                                    // [F, CP]  (K-padded)
  float* x2    = centT + (size_t)F * CP;                // [B]
  float* c2    = x2 + B;                                // [C] (rounded)
  float* S1    = c2 + CP;                               // [B, C]
  float* S2    = S1 + (size_t)B * C;                    // [B, C]
  float* P     = S2 + (size_t)B * C;                    // [B, CP] (K-padded)
  float* reach = P + (size_t)B * CP;                    // [B]
  float* mem   = reach + B;                             // [B, F]
  float* Ssel  = mem + (size_t)B * F;                   // [B, F]
  float* fused = Ssel + (size_t)B * F;                  // [B, F]

  // centroids^T (zero-padded K) for the values@centroids GEMM
  transpose_pad_kernel<<<dim3(CP / 32, F / 32), 256, 0, stream>>>(
      centroids, C, F, centT, CP);

  // squared norms
  rowsumsq_kernel<<<B, 256, 0, stream>>>(x, F, F, x2);
  rowsumsq_kernel<<<C, 256, 0, stream>>>(centroids, F, F, c2);

  // S1 = x @ centroids^T ; S2 = x @ W_hall^T + b_hall
  launch_gemm(x, F, centroids, F, nullptr, S1, C, B, C, F, stream);
  launch_gemm(x, F, W_hall,    F, b_hall,  S2, C, B, C, F, stream);

  // reachability and softmax
  min_reach_kernel<<<B, 256, 0, stream>>>(S1, C, C, x2, c2, reach);
  softmax_rows_kernel<<<B, 256, 0, stream>>>(S2, C, C, P, CP);

  // memory_feature = P @ centroids ; Ssel = x @ W_sel^T + b_sel
  launch_gemm(P, CP, centT, CP, nullptr, mem,  F, B, F, CP, stream);
  launch_gemm(x, F,  W_sel, F,  b_sel,   Ssel, F, B, F, F,  stream);

  // fuse + emit x / infused output sections
  fuse_kernel<<<(B * F + 255) / 256, 256, 0, stream>>>(
      x, Ssel, mem, reach, fused, out_x, out_inf, F, B * F);

  // logits = fused @ W_cls^T + b_cls   (dominant GEMM, W_cls streamed once/M-tile)
  launch_gemm(fused, F, W_cls, F, b_cls, logits, NCLS, B, NCLS, F, stream);
}
